// PointNetFeaturePropagation_12206297055635
// MI455X (gfx1250) — matrix operand order repaired and checked by reading, not compiled, not run on hardware
//
#include <hip/hip_runtime.h>
#include <hip/hip_bf16.h>
#include <float.h>

typedef _Float16 half_t;
typedef __attribute__((ext_vector_type(16))) half_t v16h;
typedef __attribute__((ext_vector_type(8)))  half_t v8h;
typedef __attribute__((ext_vector_type(8)))  float  v8f;

#define BN_EPS 1e-5f

static constexpr int Bb   = 8;
static constexpr int Nn   = 8192;
static constexpr int Ss   = 2048;
static constexpr int D1c  = 256;
static constexpr int D2c  = 512;
static constexpr int CIN  = 768;   // D1 + D2
static constexpr int COUT = 512;
static constexpr int Mtot = Bb * Nn;  // 65536

// CDNA5 async global->LDS copies (ASYNCcnt-tracked, no VGPR staging).
// Builtin proto (from compiler diagnostic): (v4i AS1*, v4i AS3*, imm int, imm int)
#if __has_builtin(__builtin_amdgcn_global_load_async_to_lds_b128) && \
    __has_builtin(__builtin_amdgcn_s_wait_asynccnt)
#define USE_ASYNC_LDS 1
typedef int v4i __attribute__((vector_size(16)));
typedef __attribute__((address_space(1))) v4i as1_v4i;
typedef __attribute__((address_space(3))) v4i as3_v4i;
#define AS1P(p) ((as1_v4i*)(p))
#define AS3P(p) ((as3_v4i*)(p))
#else
#define USE_ASYNC_LDS 0
#endif

// ---------------------------------------------------------------- utilities
__global__ __launch_bounds__(256) void k_zero(float* p, int n) {
    int i = blockIdx.x * 256 + threadIdx.x;
    if (i < n) p[i] = 0.0f;
}

__global__ __launch_bounds__(256) void k_f32_to_f16(const float* __restrict__ s,
                                                    half_t* __restrict__ d, int n) {
    int i = blockIdx.x * 256 + threadIdx.x;
    if (i < n) d[i] = (half_t)s[i];
}

// ------------------------------------------- 3-NN + interpolate + concat (f16)
// One wave32 per output point m = b*N + n. Produces xcat[m][0:768] f16.
__global__ __launch_bounds__(256) void k_knn_interp_concat(
    const float* __restrict__ xyz1, const float* __restrict__ xyz2,
    const float* __restrict__ p1,   const float* __restrict__ p2,
    half_t* __restrict__ xcat)
{
    const int lane = threadIdx.x & 31;
    const int wid  = threadIdx.x >> 5;
    const int m = blockIdx.x * 8 + wid;
    const int b = m >> 13;            // / 8192
    const int n = m & (Nn - 1);

    const float qx = xyz1[((size_t)b * Nn + n) * 3 + 0];
    const float qy = xyz1[((size_t)b * Nn + n) * 3 + 1];
    const float qz = xyz1[((size_t)b * Nn + n) * 3 + 2];

    float d0 = FLT_MAX, d1 = FLT_MAX, d2 = FLT_MAX;
    int   i0 = 0, i1 = 0, i2 = 0;
    const float* x2 = xyz2 + (size_t)b * Ss * 3;
    for (int s = lane; s < Ss; s += 32) {
        float dx = qx - x2[s * 3 + 0];
        float dy = qy - x2[s * 3 + 1];
        float dz = qz - x2[s * 3 + 2];
        float d  = dx * dx + dy * dy + dz * dz;
        if (d < d2) {
            if (d < d1) {
                d2 = d1; i2 = i1;
                if (d < d0) { d1 = d0; i1 = i0; d0 = d; i0 = s; }
                else        { d1 = d;  i1 = s; }
            } else { d2 = d; i2 = s; }
        }
    }

    // merge per-lane sorted top-3 across the wave: 3 rounds of min-reduce+pop
    float wd[3]; int wi[3];
    for (int r = 0; r < 3; ++r) {
        float bd = d0; int bi = i0; int bl = lane;
        for (int off = 16; off > 0; off >>= 1) {
            float od = __shfl_xor(bd, off);
            int   oi = __shfl_xor(bi, off);
            int   ol = __shfl_xor(bl, off);
            if (od < bd || (od == bd && ol < bl)) { bd = od; bi = oi; bl = ol; }
        }
        wd[r] = bd; wi[r] = bi;
        if (lane == bl) { d0 = d1; i0 = i1; d1 = d2; i1 = i2; d2 = FLT_MAX; }
    }

    const float r0 = 1.0f / (wd[0] + 1e-8f);
    const float r1 = 1.0f / (wd[1] + 1e-8f);
    const float r2 = 1.0f / (wd[2] + 1e-8f);
    const float rs = 1.0f / (r0 + r1 + r2);
    const float w0 = r0 * rs, w1 = r1 * rs, w2 = r2 * rs;

    half_t* row = xcat + (size_t)m * CIN;
    const float* p1b = p1 + (size_t)b * D1c * Nn;   // [D1][N]
    for (int c = lane; c < D1c; c += 32)
        row[c] = (half_t)p1b[(size_t)c * Nn + n];
    const float* p2b = p2 + (size_t)b * D2c * Ss;   // [D2][S]
    for (int c = lane; c < D2c; c += 32) {
        const float* pc = p2b + (size_t)c * Ss;
        float v = w0 * pc[wi[0]] + w1 * pc[wi[1]] + w2 * pc[wi[2]];
        row[D1c + c] = (half_t)v;
    }
}

// --------------------------------------------------- WMMA GEMM + BN statistics
// Y[m][n] = sum_k A[m][k] * Wt[n][k] + bias[n]   (x @ W^T + b)
// Block tile 128(M) x 64(N), 8 waves -> each wave 32x32 (2x2 WMMA 16x16x32 f16).
// Double-buffered LDS; stage k+1 streamed in with async global->LDS copies
// (3 x b128 per thread per stage) while stage k feeds the WMMA pipes.
union FragH { v16h v; v8h h[2]; };

__global__ __launch_bounds__(256) void k_gemm_stats(
    const half_t* __restrict__ A,    // [M, K] row-major f16
    const half_t* __restrict__ Wt,   // [N, K] row-major f16 (== W as given)
    const float*  __restrict__ bias, // [N]
    half_t* __restrict__ Y,          // [M, N] f16
    float* __restrict__ csum, float* __restrict__ csumsq,  // [N] each
    int M, int N, int K)
{
    __shared__ alignas(16) half_t sA[2][128 * 40];  // [row][k], stride 40 halfs
    __shared__ alignas(16) half_t sB[2][64 * 40];   // [n][k],   stride 40 halfs

    const int tid   = threadIdx.x;
    const int lane  = tid & 31;
    const int wid   = tid >> 5;
    const int waveM = wid & 3;              // 32-row slice
    const int waveN = wid >> 2;             // 32-col slice
    const int mBase = blockIdx.y * 128;
    const int nBase = blockIdx.x * 64;

    const int grp = lane >> 4;              // lane group (0: lanes 0-15, 1: 16-31)
    const int rr  = lane & 15;

    const v8f vzero = {0.f, 0.f, 0.f, 0.f, 0.f, 0.f, 0.f, 0.f};
    v8f acc[4] = {vzero, vzero, vzero, vzero};

    const int ar = tid >> 1, ac = tid & 1;  // A fill: 128 rows x 2 chunks of 16 halfs
    const int bn = tid >> 2, bc = tid & 3;  // B fill: 64 rows  x 4 chunks of 8 halfs

    // Issue the global->LDS copies for one K-stage into buffer p.
    auto issue_stage = [&](int p, int kk) {
        const half_t* ga = A  + (size_t)(mBase + ar) * K + kk + ac * 16;
        const half_t* gb = Wt + (size_t)(nBase + bn) * K + kk + bc * 8;
        half_t* la = &sA[p][ar * 40 + ac * 16];
        half_t* lb = &sB[p][bn * 40 + bc * 8];
#if USE_ASYNC_LDS
        __builtin_amdgcn_global_load_async_to_lds_b128(AS1P(ga),     AS3P(la),     0, 0);
        __builtin_amdgcn_global_load_async_to_lds_b128(AS1P(ga + 8), AS3P(la + 8), 0, 0);
        __builtin_amdgcn_global_load_async_to_lds_b128(AS1P(gb),     AS3P(lb),     0, 0);
#else
        float4 va0 = ((const float4*)ga)[0];
        float4 va1 = ((const float4*)ga)[1];
        float4 vb  = *(const float4*)gb;
        ((float4*)la)[0] = va0;
        ((float4*)la)[1] = va1;
        *(float4*)lb = vb;
#endif
    };

    const int nsteps = K / 32;
    issue_stage(0, 0);                      // prologue: stage 0 in flight

    for (int it = 0; it < nsteps; ++it) {
        const int p = it & 1;
        if (it + 1 < nsteps) {
            issue_stage(p ^ 1, (it + 1) * 32);   // overlap next stage with compute
#if USE_ASYNC_LDS
            __builtin_amdgcn_s_wait_asynccnt(3); // in-order: stage it complete
#endif
        } else {
#if USE_ASYNC_LDS
            __builtin_amdgcn_s_wait_asynccnt(0);
#endif
        }
        __syncthreads();                    // stage p visible to all waves

        // A fragment layout (16-bit, 16x32): lane holds M = lane%16,
        // K = (e/8)*16 + grp*8 + e%8  -> two contiguous 8-half chunks.
        FragH a0, a1, b0, b1;
        const half_t* ap0 = &sA[p][(waveM * 32 + rr) * 40];
        const half_t* ap1 = &sA[p][(waveM * 32 + 16 + rr) * 40];
        a0.h[0] = *(const v8h*)(ap0 + grp * 8);
        a0.h[1] = *(const v8h*)(ap0 + 16 + grp * 8);
        a1.h[0] = *(const v8h*)(ap1 + grp * 8);
        a1.h[1] = *(const v8h*)(ap1 + 16 + grp * 8);
        // B fragment (32x16): lane holds N = lane%16, K = (e/8)*16 + grp*8 + e%8 (same as A).
        const half_t* bp0 = &sB[p][(waveN * 32 + rr) * 40];
        const half_t* bp1 = &sB[p][(waveN * 32 + 16 + rr) * 40];
        b0.h[0] = *(const v8h*)(bp0 + grp * 8);  b0.h[1] = *(const v8h*)(bp0 + 16 + grp * 8);
        b1.h[0] = *(const v8h*)(bp1 + grp * 8);  b1.h[1] = *(const v8h*)(bp1 + 16 + grp * 8);

        acc[0] = __builtin_amdgcn_wmma_f32_16x16x32_f16(false, a0.v, false, b0.v, (short)0, acc[0], false, false);
        acc[1] = __builtin_amdgcn_wmma_f32_16x16x32_f16(false, a0.v, false, b1.v, (short)0, acc[1], false, false);
        acc[2] = __builtin_amdgcn_wmma_f32_16x16x32_f16(false, a1.v, false, b0.v, (short)0, acc[2], false, false);
        acc[3] = __builtin_amdgcn_wmma_f32_16x16x32_f16(false, a1.v, false, b1.v, (short)0, acc[3], false, false);

        __syncthreads();                    // buffer p free for reuse next iter
    }

    // Epilogue: +bias, per-channel sum / sumsq via global f32 atomics, f16 store.
    // C/D layout: VGPR v, lane: N = lane%16, M = v + (lane>=16 ? 8 : 0).
    #pragma unroll
    for (int t = 0; t < 4; ++t) {
        const int tm = (t >> 1) * 16;
        const int tn = (t & 1) * 16;
        const int n    = nBase + waveN * 32 + tn + rr;
        const int mrow = mBase + waveM * 32 + tm + grp * 8;
        const float bv = bias[n];
        float bsum = 0.f, bsq = 0.f;
        #pragma unroll
        for (int v = 0; v < 8; ++v) {
            float val = acc[t][v] + bv;
            bsum += val;
            bsq  += val * val;
            Y[(size_t)(mrow + v) * N + n] = (half_t)val;
        }
        // lanes L and L+16 share channel n -> fold before atomics
        float osum = __shfl_down(bsum, 16);
        float osq  = __shfl_down(bsq, 16);
        if (lane < 16) {
            atomicAdd(&csum[n],   bsum + osum);
            atomicAdd(&csumsq[n], bsq + osq);
        }
    }
}

// ------------------------------------------------------------- BN + ReLU (f16)
__global__ __launch_bounds__(256) void k_bn_relu(
    const half_t* __restrict__ Y,
    const float* __restrict__ csum, const float* __restrict__ csumsq,
    const float* __restrict__ g, const float* __restrict__ be,
    half_t* __restrict__ X, int total, float invM)
{
    int i = blockIdx.x * 256 + threadIdx.x;
    if (i >= total) return;
    int c = i & (COUT - 1);
    float mean = csum[c] * invM;
    float var  = csumsq[c] * invM - mean * mean;
    float val  = ((float)Y[i] - mean) * rsqrtf(var + BN_EPS) * g[c] + be[c];
    X[i] = (half_t)(val > 0.f ? val : 0.f);
}

// --------------------------- final BN + residual + ReLU + transpose to [B,C,N]
__global__ __launch_bounds__(256) void k_final(
    const half_t* __restrict__ Y2, const half_t* __restrict__ X1,
    const float* __restrict__ csum, const float* __restrict__ csumsq,
    const float* __restrict__ g, const float* __restrict__ be,
    float* __restrict__ out, float invM)
{
    const int n = blockIdx.x * 256 + threadIdx.x;
    const int c = blockIdx.y;
    const int b = blockIdx.z;
    const size_t m = (size_t)b * Nn + n;
    const size_t i = m * COUT + c;
    float mean = csum[c] * invM;
    float var  = csumsq[c] * invM - mean * mean;
    float val  = ((float)Y2[i] - mean) * rsqrtf(var + BN_EPS) * g[c] + be[c] + (float)X1[i];
    out[((size_t)b * COUT + c) * Nn + n] = val > 0.f ? val : 0.f;
}

// -------------------------------------------------------------------- launcher
extern "C" void kernel_launch(void* const* d_in, const int* in_sizes, int n_in,
                              void* d_out, int out_size, void* d_ws, size_t ws_size,
                              hipStream_t stream)
{
    (void)in_sizes; (void)n_in; (void)out_size; (void)ws_size;
    const float* xyz1    = (const float*)d_in[0];
    const float* xyz2    = (const float*)d_in[1];
    const float* points1 = (const float*)d_in[2];
    const float* points2 = (const float*)d_in[3];
    const float* W_fuse  = (const float*)d_in[4];
    const float* b_fuse  = (const float*)d_in[5];
    const float* g_fuse  = (const float*)d_in[6];
    const float* be_fuse = (const float*)d_in[7];
    const float* W1      = (const float*)d_in[8];
    const float* b1      = (const float*)d_in[9];
    const float* g1      = (const float*)d_in[10];
    const float* be1     = (const float*)d_in[11];
    const float* W2      = (const float*)d_in[12];
    const float* b2      = (const float*)d_in[13];
    const float* g2      = (const float*)d_in[14];
    const float* be2     = (const float*)d_in[15];
    float* out = (float*)d_out;

    // workspace carve-up (256B aligned slabs)
    char* ws = (char*)d_ws;
    size_t off = 0;
    auto take = [&](size_t bytes) {
        char* p = ws + off;
        off = (off + bytes + 255) & ~(size_t)255;
        return p;
    };
    half_t* xcat  = (half_t*)take((size_t)Mtot * CIN  * sizeof(half_t)); // reused as h
    half_t* ybuf  = (half_t*)take((size_t)Mtot * COUT * sizeof(half_t));
    half_t* x1    = (half_t*)take((size_t)Mtot * COUT * sizeof(half_t));
    half_t* Wf_h  = (half_t*)take((size_t)CIN  * COUT * sizeof(half_t));
    half_t* W1_h  = (half_t*)take((size_t)COUT * COUT * sizeof(half_t));
    half_t* W2_h  = (half_t*)take((size_t)COUT * COUT * sizeof(half_t));
    float*  stats = (float*)take(6 * COUT * sizeof(float));

    // 1) zero BN statistic accumulators (must be per-launch for determinism)
    k_zero<<<(6 * COUT + 255) / 256, 256, 0, stream>>>(stats, 6 * COUT);

    // 2) weights f32 -> f16
    k_f32_to_f16<<<(CIN * COUT + 255) / 256, 256, 0, stream>>>(W_fuse, Wf_h, CIN * COUT);
    k_f32_to_f16<<<(COUT * COUT + 255) / 256, 256, 0, stream>>>(W1, W1_h, COUT * COUT);
    k_f32_to_f16<<<(COUT * COUT + 255) / 256, 256, 0, stream>>>(W2, W2_h, COUT * COUT);

    // 3) 3-NN interpolation + concat -> xcat [M, 768] f16
    k_knn_interp_concat<<<Mtot / 8, 256, 0, stream>>>(xyz1, xyz2, points1, points2, xcat);

    const float invM = 1.0f / (float)Mtot;
    const dim3 gg(COUT / 64, Mtot / 128);
    const int  elt_blocks = (Mtot * COUT) / 256;

    // 4) fuse GEMM (768 -> 512) + stats, then BN+ReLU -> x1
    k_gemm_stats<<<gg, 256, 0, stream>>>(xcat, Wf_h, b_fuse, ybuf,
                                         stats + 0, stats + COUT, Mtot, COUT, CIN);
    k_bn_relu<<<elt_blocks, 256, 0, stream>>>(ybuf, stats + 0, stats + COUT,
                                              g_fuse, be_fuse, x1, Mtot * COUT, invM);

    // 5) res-block first GEMM + BN+ReLU -> h (reuse xcat slab)
    half_t* hbuf = xcat;
    k_gemm_stats<<<gg, 256, 0, stream>>>(x1, W1_h, b1, ybuf,
                                         stats + 2 * COUT, stats + 3 * COUT, Mtot, COUT, COUT);
    k_bn_relu<<<elt_blocks, 256, 0, stream>>>(ybuf, stats + 2 * COUT, stats + 3 * COUT,
                                              g1, be1, hbuf, Mtot * COUT, invM);

    // 6) res-block second GEMM, then BN + residual + ReLU + transpose -> out
    k_gemm_stats<<<gg, 256, 0, stream>>>(hbuf, W2_h, b2, ybuf,
                                         stats + 4 * COUT, stats + 5 * COUT, Mtot, COUT, COUT);
    k_final<<<dim3(Nn / 256, COUT, Bb), 256, 0, stream>>>(ybuf, x1,
                                                          stats + 4 * COUT, stats + 5 * COUT,
                                                          g2, be2, out, invM);
}
